// FlashLlamaAttention_60662118089380
// MI455X (gfx1250) — matrix operand order repaired and checked by reading, not compiled, not run on hardware
//
#include <hip/hip_runtime.h>
#include <math.h>

#define BATCH 32
#define HID   4096
#define NH    32
#define NKV   8
#define HS    128
#define MAXB  128   // 2048 / 16

typedef __attribute__((ext_vector_type(16))) __bf16 v16bf;
typedef __attribute__((ext_vector_type(8)))  float  v8f;
typedef __attribute__((ext_vector_type(4)))  float  v4f;

// gfx1250 async global->LDS copy (ASYNCcnt-tracked).  vdst VGPR holds the LDS
// byte offset (low 32 bits of the generic shared-memory address per the ISA
// aperture mapping), vaddr is the 64-bit global address pair.
__device__ __forceinline__ void async_load_lds_b128(void* lds_ptr, const void* gptr) {
    unsigned loff = (unsigned)(unsigned long long)lds_ptr;
    asm volatile("global_load_async_to_lds_b128 %0, %1, off"
                 :: "v"(loff), "v"(gptr) : "memory");
}
__device__ __forceinline__ void wait_asynccnt0() {
    asm volatile("s_wait_asynccnt 0x0" ::: "memory");
}

// ---------------------------------------------------------------------------
// Generic  Y[col, row] = sum_k W[row, k] * X[col, k]   with K = HID = 4096,
// 32 columns (batch), rows = gridDim.x * 16.  One wave per 16-row tile,
// two 16x16 accumulators covering all 32 batch columns so each weight element
// is read exactly once (bandwidth-optimal).  bf16 WMMA, fp32 accumulate.
// ---------------------------------------------------------------------------
__global__ __launch_bounds__(32) void gemm_wmma(const float* __restrict__ W,
                                                const float* __restrict__ X,
                                                float* __restrict__ Y,
                                                int ldy) {
    const int lane = threadIdx.x & 31;
    const int half = lane >> 4;                 // 0: low 16 lanes, 1: high
    const int arow = lane & 15;
    const float* wrow = W + (size_t)(blockIdx.x * 16 + arow) * HID;
    const float* x0   = X + (size_t)arow * HID;          // batch cols 0..15
    const float* x1   = x0 + (size_t)16 * HID;           // batch cols 16..31

    v8f acc0 = {}; v8f acc1 = {};

    for (int k0 = 0; k0 < HID; k0 += 32) {
        __builtin_prefetch(wrow + k0 + 256, 0, 0);       // global_prefetch_b8
        // ---- A tile: 16 rows x 32 K (bf16) ----
        // lane<16: K = k0 + 0..15;  lane>=16: K = k0 + 16..31
        const int ka = k0 + half * 16;
        v4f w0 = *(const v4f*)(wrow + ka);
        v4f w1 = *(const v4f*)(wrow + ka + 4);
        v4f w2 = *(const v4f*)(wrow + ka + 8);
        v4f w3 = *(const v4f*)(wrow + ka + 12);
        v16bf a;
        #pragma unroll
        for (int e = 0; e < 4; ++e) {
            a[e]      = (__bf16)w0[e];
            a[e + 4]  = (__bf16)w1[e];
            a[e + 8]  = (__bf16)w2[e];
            a[e + 12] = (__bf16)w3[e];
        }
        // ---- B tiles: 32 K x 16 cols (bf16) ----
        // lane<16: K = k0+0..15 for col=lane;  lane>=16: K = k0+16..31
        const int kb = k0 + half * 16;
        v16bf b0, b1;
        #pragma unroll
        for (int e = 0; e < 16; e += 4) {
            v4f t0 = *(const v4f*)(x0 + kb + e);
            v4f t1 = *(const v4f*)(x1 + kb + e);
            #pragma unroll
            for (int j = 0; j < 4; ++j) {
                b0[e + j] = (__bf16)t0[j];
                b1[e + j] = (__bf16)t1[j];
            }
        }
        acc0 = __builtin_amdgcn_wmma_f32_16x16x32_bf16(false, a, false, b0,
                                                       (short)0, acc0, false, false);
        acc1 = __builtin_amdgcn_wmma_f32_16x16x32_bf16(false, a, false, b1,
                                                       (short)0, acc1, false, false);
    }
    // D layout: VGPR i -> M = i + 8*half, N = lane&15.  Rows are contiguous in
    // Y, so store each v8f accumulator as two b128s.
    const int mbase = blockIdx.x * 16 + (half ? 8 : 0);
    const int col   = lane & 15;
    float* yp0 = Y + (size_t)col        * ldy + mbase;
    float* yp1 = Y + (size_t)(col + 16) * ldy + mbase;
    *(v4f*)(yp0)     = __builtin_shufflevector(acc0, acc0, 0, 1, 2, 3);
    *(v4f*)(yp0 + 4) = __builtin_shufflevector(acc0, acc0, 4, 5, 6, 7);
    *(v4f*)(yp1)     = __builtin_shufflevector(acc1, acc1, 0, 1, 2, 3);
    *(v4f*)(yp1 + 4) = __builtin_shufflevector(acc1, acc1, 4, 5, 6, 7);
}

// ---------------------------------------------------------------------------
// In-place RoPE on q (B,NH,HS) and k (B,NKV,HS); position = len-1.
// ---------------------------------------------------------------------------
__global__ void rope_kernel(float* __restrict__ q, float* __restrict__ k,
                            const int* __restrict__ lens) {
    const int total = BATCH * (NH + NKV) * 64;
    int idx = blockIdx.x * blockDim.x + threadIdx.x;
    if (idx >= total) return;
    int j  = idx & 63;
    int t  = idx >> 6;
    int hh = t % (NH + NKV);
    int b  = t / (NH + NKV);
    float pos = (float)(lens[b] - 1);
    float inv = powf(10000.0f, -(float)(2 * j) / 128.0f);
    float f = pos * inv;
    float c, s;
    __sincosf(f, &s, &c);
    float* base = (hh < NH) ? (q + ((size_t)b * NH + hh) * HS)
                            : (k + ((size_t)b * NKV + (hh - NH)) * HS);
    float x1 = base[j], x2 = base[j + 64];
    base[j]      = x1 * c - x2 * s;
    base[j + 64] = x2 * c + x1 * s;
}

// ---------------------------------------------------------------------------
// Paged flash-decode attention: one workgroup (4 waves) per (batch, kv-head).
// Per 16-token page: (1) async-stream the V page into LDS, (2) QK^T via 4
// chained bf16 WMMAs (overlapped with the async V stream), (3) wave-uniform
// online softmax, (4) s_wait_asynccnt then PV from LDS.  Position len-1
// substitutes the fresh k/v rows (virtual cache-slot update).  Cross-wave
// combine in LDS.
// ---------------------------------------------------------------------------
__global__ __launch_bounds__(128) void attn_kernel(
    const float* __restrict__ kc, const float* __restrict__ vc,
    const float* __restrict__ qws, const float* __restrict__ kws,
    const float* __restrict__ vws, const int* __restrict__ btab,
    const int* __restrict__ lens, float* __restrict__ aout) {

    __shared__ float vlds[4][16][HS];      // per-wave V page staging (32 KB)
    __shared__ float sm[4][4];
    __shared__ float sl[4][4];
    __shared__ float sacc[4][4][HS];

    const int b = blockIdx.x / NKV;
    const int h = blockIdx.x % NKV;
    const int wave = threadIdx.x >> 5;
    const int lane = threadIdx.x & 31;
    const int half = lane >> 4;
    const int key  = lane & 15;
    const int len  = lens[b];
    const int last = len - 1;
    const int npages = (len + 15) >> 4;
    const float scale = 0.08838834764831845f;   // 1/sqrt(128)

    const float* knew = kws + ((size_t)b * NKV + h) * HS;
    const float* vnew = vws + ((size_t)b * NKV + h) * HS;

    // ---- preload q as 4 A-matrix chunks (rows 0..3 valid, 4..15 zero) ----
    const int arow = lane & 15;
    const float* qrow = qws + ((size_t)b * NH + h * 4 + (arow < 4 ? arow : 0)) * HS;
    const v16bf zero16 = {};
    v16bf aq[4];
    #pragma unroll
    for (int c = 0; c < 4; ++c) {
        const int ka = c * 32 + half * 16;
        v4f w0 = *(const v4f*)(qrow + ka);
        v4f w1 = *(const v4f*)(qrow + ka + 4);
        v4f w2 = *(const v4f*)(qrow + ka + 8);
        v4f w3 = *(const v4f*)(qrow + ka + 12);
        v16bf a;
        #pragma unroll
        for (int e = 0; e < 4; ++e) {
            a[e]      = (__bf16)w0[e];
            a[e + 4]  = (__bf16)w1[e];
            a[e + 8]  = (__bf16)w2[e];
            a[e + 12] = (__bf16)w3[e];
        }
        aq[c] = (arow < 4) ? a : zero16;
    }

    float mst[4], lst[4], acc[4][4];
    #pragma unroll
    for (int m = 0; m < 4; ++m) {
        mst[m] = -1e30f; lst[m] = 0.0f;
        acc[m][0] = acc[m][1] = acc[m][2] = acc[m][3] = 0.0f;
    }

    for (int page = wave; page < npages; page += 4) {
        const int blk = btab[b * MAXB + page];

        // ---- (1) async-stream V page into this wave's LDS buffer ----
        #pragma unroll
        for (int kk = 0; kk < 16; ++kk) {
            const int kpos = page * 16 + kk;
            const float* vrow = (kpos == last)
                ? vnew : (vc + (((size_t)(blk * 16 + kk)) * NKV + h) * HS);
            async_load_lds_b128(&vlds[wave][kk][lane * 4], vrow + lane * 4);
        }

        // ---- (2) scores = Q . K^T via 4 WMMAs over K=128 (overlaps async V) ----
        const int pos = page * 16 + key;
        const float* krow = (pos == last)
            ? knew : (kc + (((size_t)(blk * 16 + key)) * NKV + h) * HS);
        v8f sc = {};
        #pragma unroll
        for (int c = 0; c < 4; ++c) {
            const int kb = c * 32 + half * 16;
            v16bf bm;
            #pragma unroll
            for (int e = 0; e < 16; e += 4) {
                v4f t = *(const v4f*)(krow + kb + e);
                #pragma unroll
                for (int j = 0; j < 4; ++j) bm[e + j] = (__bf16)t[j];
            }
            sc = __builtin_amdgcn_wmma_f32_16x16x32_bf16(false, aq[c], false, bm,
                                                         (short)0, sc, false, false);
        }

        // ---- (3) online softmax (wave-uniform state; upper half masked) ----
        const bool valid = (half == 0) && (pos < len);
        float p[4];
        #pragma unroll
        for (int m = 0; m < 4; ++m) {
            float s = valid ? sc[m] * scale : -1e30f;
            float rmax = s;
            #pragma unroll
            for (int off = 16; off; off >>= 1)
                rmax = fmaxf(rmax, __shfl_xor(rmax, off, 32));
            float mnew = fmaxf(mst[m], rmax);
            float corr = __expf(mst[m] - mnew);
            float pe   = __expf(s - mnew);
            float rsum = pe;
            #pragma unroll
            for (int off = 16; off; off >>= 1)
                rsum += __shfl_xor(rsum, off, 32);
            lst[m] = lst[m] * corr + rsum;
            mst[m] = mnew;
            p[m]   = pe;
            acc[m][0] *= corr; acc[m][1] *= corr;
            acc[m][2] *= corr; acc[m][3] *= corr;
        }

        // ---- (4) PV from LDS: each lane owns dims [lane*4, lane*4+4) ----
        wait_asynccnt0();
        const int nk = min(16, len - page * 16);
        for (int kk = 0; kk < nk; ++kk) {
            v4f vv = *(const v4f*)&vlds[wave][kk][lane * 4];
            #pragma unroll
            for (int m = 0; m < 4; ++m) {
                float pm = __shfl(p[m], kk, 32);
                acc[m][0] += pm * vv[0];
                acc[m][1] += pm * vv[1];
                acc[m][2] += pm * vv[2];
                acc[m][3] += pm * vv[3];
            }
        }
    }

    // ---- cross-wave combine in LDS ----
    #pragma unroll
    for (int m = 0; m < 4; ++m) {
        #pragma unroll
        for (int j = 0; j < 4; ++j) sacc[wave][m][lane * 4 + j] = acc[m][j];
        if (lane == 0) { sm[wave][m] = mst[m]; sl[wave][m] = lst[m]; }
    }
    __syncthreads();

    const int d = threadIdx.x;   // 0..127
    #pragma unroll
    for (int m = 0; m < 4; ++m) {
        float gm = fmaxf(fmaxf(sm[0][m], sm[1][m]), fmaxf(sm[2][m], sm[3][m]));
        float gl = 0.0f, val = 0.0f;
        #pragma unroll
        for (int w = 0; w < 4; ++w) {
            float f = __expf(sm[w][m] - gm);
            gl  += sl[w][m] * f;
            val += sacc[w][m][d] * f;
        }
        aout[((size_t)b * NH + h * 4 + m) * HS + d] = val / gl;
    }
}

// ---------------------------------------------------------------------------
extern "C" void kernel_launch(void* const* d_in, const int* in_sizes, int n_in,
                              void* d_out, int out_size, void* d_ws, size_t ws_size,
                              hipStream_t stream) {
    const float* hs   = (const float*)d_in[0];
    const float* wq   = (const float*)d_in[1];
    const float* wk   = (const float*)d_in[2];
    const float* wv   = (const float*)d_in[3];
    const float* wo   = (const float*)d_in[4];
    const float* kc   = (const float*)d_in[5];
    const float* vc   = (const float*)d_in[6];
    const int*   btab = (const int*)d_in[8];
    const int*   lens = (const int*)d_in[10];
    float*       out  = (float*)d_out;

    float* qws = (float*)d_ws;                      // B * NH  * HS = 131072
    float* kws = qws + (size_t)BATCH * NH * HS;     // B * NKV * HS =  32768
    float* vws = kws + (size_t)BATCH * NKV * HS;    // B * NKV * HS =  32768
    float* aws = vws + (size_t)BATCH * NKV * HS;    // B * NH  * HS = 131072

    // QKV projections (weight-streaming bf16-WMMA GEMMs)
    gemm_wmma<<<(NH  * HS) / 16, 32, 0, stream>>>(wq, hs, qws, NH * HS);
    gemm_wmma<<<(NKV * HS) / 16, 32, 0, stream>>>(wk, hs, kws, NKV * HS);
    gemm_wmma<<<(NKV * HS) / 16, 32, 0, stream>>>(wv, hs, vws, NKV * HS);

    // RoPE on q and k
    const int rope_items = BATCH * (NH + NKV) * 64;
    rope_kernel<<<(rope_items + 255) / 256, 256, 0, stream>>>(qws, kws, lens);

    // Paged flash-decode attention
    attn_kernel<<<BATCH * NKV, 128, 0, stream>>>(kc, vc, qws, kws, vws,
                                                 btab, lens, aws);

    // Output projection -> d_out (B, HID)
    gemm_wmma<<<HID / 16, 32, 0, stream>>>(wo, aws, out, HID);
}
